// DiffPoolModel_37598143709438
// MI455X (gfx1250) — matrix operand order repaired and checked by reading, not compiled, not run on hardware
//
#include <hip/hip_runtime.h>
#include <hip/hip_bf16.h>

typedef __bf16 bf16;
typedef __attribute__((ext_vector_type(16))) __bf16 v16bf;
typedef __attribute__((ext_vector_type(8)))  float  v8f;

#define NTOT 262144   // B*64 nodes
#define BGR  4096     // graphs
#define EPG  512      // edges per graph

union FragU { v16bf h; unsigned int u[8]; };
union PairU { bf16 h[2]; unsigned int u; };

static __device__ __forceinline__ unsigned int pk2(float a, float b){
  PairU t; t.h[0] = (bf16)a; t.h[1] = (bf16)b; return t.u;   // -> v_cvt_pk_bf16_f32
}

// ---- WMMA fragment loaders (CDNA5 16-bit layouts, wave32) -------------------
// A: 16x32 tile (MxK), row-major, element (m,k) at A[m*ld + k]; K-pairs contiguous
// lane l: m = l&15 ; kbase = (l>>4)*8 ; vgpr v holds K pair (v<4 ? 2v : 16+2(v-4)) + kbase
static __device__ __forceinline__ v16bf load_a(const bf16* __restrict__ A, int ld, int lane){
  int m  = lane & 15;
  int kb = (lane >> 4) << 3;
  const unsigned int* row = (const unsigned int*)(A + (size_t)m*ld);  // ld even, base 4B-aligned
  FragU f;
  #pragma unroll
  for (int v = 0; v < 8; ++v){
    int k = ((v < 4) ? (2*v) : (16 + 2*(v-4))) + kb;
    f.u[v] = row[k >> 1];
  }
  return f.h;
}

// A from fp32 source (x input): element loads + convert (compiler packs with cvt_pk)
static __device__ __forceinline__ v16bf load_a(const float* __restrict__ A, int ld, int lane){
  int m  = lane & 15;
  int kb = (lane >> 4) << 3;
  v16bf f;
  #pragma unroll
  for (int v = 0; v < 8; ++v){
    int k = ((v < 4) ? (2*v) : (16 + 2*(v-4))) + kb;
    f[2*v]   = (bf16)A[(size_t)m*ld + k];
    f[2*v+1] = (bf16)A[(size_t)m*ld + k + 1];
  }
  return f;
}

// B: 32x16 tile (KxN) from K-pair-interleaved storage: uint P[(k>>1)*F + n] = {B[k][n], B[k+1][n]}
// lane l: n = l&15 ; kb2 = (l>=16)?4:0 ; vgpr v holds pair (v<4 ? v : 4+v) + kb2
static __device__ __forceinline__ v16bf load_b_pk(const unsigned int* __restrict__ P, int F, int lane){
  int n   = lane & 15;
  int kb2 = (lane >> 4) << 2;
  FragU f;
  #pragma unroll
  for (int v = 0; v < 8; ++v)
    f.u[v] = P[(size_t)(((v < 4) ? v : (4 + v)) + kb2)*F + n];
  return f.h;
}

static __device__ __forceinline__ v8f wmma_bf(v16bf a, v16bf b, v8f c){
  return __builtin_amdgcn_wmma_f32_16x16x32_bf16(false, a, false, b, (short)0, c, false, false);
}

// ---- kernels ----------------------------------------------------------------

__global__ void k_zero(float* p, int n){
  int i = blockIdx.x * blockDim.x + threadIdx.x;
  if (i < n) p[i] = 0.0f;
}

// Pack fp32 weights W[K x Fv] -> interleaved bf16 pairs Wp[(k>>1)*Fpad + n], zero-padded cols
__global__ void k_packW(const float* __restrict__ W, unsigned int* __restrict__ Wp,
                        int K, int Fv, int Fpad){
  int i = blockIdx.x * blockDim.x + threadIdx.x;
  int tot = (K >> 1) * Fpad;
  if (i >= tot) return;
  int kp = i / Fpad, n = i % Fpad;
  float a = (n < Fv) ? W[(size_t)(2*kp)    *Fv + n] : 0.0f;
  float b = (n < Fv) ? W[(size_t)(2*kp + 1)*Fv + n] : 0.0f;
  Wp[i] = pk2(a, b);
}

// Per-graph: edge counts in LDS, degree+dis; emit interleaved adjcnt (B-operand layout
// over first index) and row-major Ahat[i][j] = dis_i*dis_j*(A[j][i] + (i==j))
__global__ void k_adj(const int* __restrict__ src, const int* __restrict__ dst,
                      bf16* __restrict__ adjcnt, bf16* __restrict__ ahat){
  int g = blockIdx.x, tid = threadIdx.x;
  __shared__ float cnt[64*64];
  __shared__ float dis[64];
  for (int i = tid; i < 4096; i += 256) cnt[i] = 0.0f;
  __syncthreads();
  const int* S = src + (size_t)g*EPG;
  const int* D = dst + (size_t)g*EPG;
  for (int e = tid; e < EPG; e += 256){
    int ps = S[e] & 63, pd = D[e] & 63;
    atomicAdd(&cnt[ps*64 + pd], 1.0f);
  }
  __syncthreads();
  if (tid < 64){
    float dg = 1.0f;
    for (int s = 0; s < 64; ++s) dg += cnt[s*64 + tid];
    dis[tid] = rsqrtf(dg);
  }
  __syncthreads();
  bf16* AC = adjcnt + (size_t)g*4096;
  bf16* AH = ahat   + (size_t)g*4096;
  for (int idx = tid; idx < 4096; idx += 256){
    int i = idx >> 6, j = idx & 63;
    // adjcnt interleaved over first (K) index: A[i][j] -> [(i>>1)*128 + 2j + (i&1)]
    AC[((i >> 1)*128) + 2*j + (i & 1)] = (bf16)cnt[idx];
    float a = cnt[j*64 + i] + (i == j ? 1.0f : 0.0f);
    AH[idx] = (bf16)(dis[i]*dis[j]*a);
  }
}

// Cpk[(row>>1)*Fpad + col] (interleaved pairs) = A[Mtot x K] @ Wpk (pre-packed)
template<typename TA>
__global__ void k_gemm_flat(const TA* __restrict__ A, const unsigned int* __restrict__ Wpk,
                            unsigned int* __restrict__ Cpk, int K, int Fpad){
  int lane = threadIdx.x & 31, wave = threadIdx.x >> 5;
  int mt = blockIdx.x * (blockDim.x >> 5) + wave;
  int nt = blockIdx.y;
  const TA* At = A + (size_t)mt * 16 * K;
  const unsigned int* Wt = Wpk + nt*16;
  v8f c = {};
  for (int k = 0; k < K; k += 32){
    v16bf a = load_a(At + k, K, lane);
    v16bf b = load_b_pk(Wt + (size_t)(k >> 1)*Fpad, Fpad, lane);
    c = wmma_bf(a, b, c);
  }
  int n = lane & 15, mb = (lane >> 4) << 3;
  int cg = nt*16 + n;
  unsigned int* Ct = Cpk + (size_t)mt*8*Fpad + cg;     // mt*16 rows = mt*8 pairs
  #pragma unroll
  for (int r = 0; r < 8; r += 2)
    Ct[(size_t)((mb + r) >> 1)*Fpad] = pk2(c[r], c[r+1]);
}

// Out_g[M x F] = act(Ahat_g[M x K] @ G_g + bias); G pre-packed interleaved.
// OUTMODE: 0 = row-major bf16, 1 = interleaved-pair bf16 (B-operand layout), 2 = fp32
template<bool RELU, int OUTMODE>
__global__ void k_graph_agg(const bf16* __restrict__ Ah, const unsigned int* __restrict__ Gpk,
                            const float* __restrict__ bias, int nbias,
                            bf16* __restrict__ Obf, unsigned int* __restrict__ Opk,
                            float* __restrict__ Of, int M, int K, int F){
  int g = blockIdx.x, nt = blockIdx.y;
  int lane = threadIdx.x & 31, wave = threadIdx.x >> 5;   // wave = row tile
  const bf16* At = Ah + (size_t)g*M*K + (size_t)wave*16*K;
  const unsigned int* Gt = Gpk + (size_t)g*(K >> 1)*F + nt*16;
  v8f c = {};
  for (int k = 0; k < K; k += 32){
    v16bf a = load_a(At + k, K, lane);
    v16bf b = load_b_pk(Gt + (size_t)(k >> 1)*F, F, lane);
    c = wmma_bf(a, b, c);
  }
  int n = lane & 15, mb = (lane >> 4) << 3;
  int cg = nt*16 + n;
  float bv = (cg < nbias) ? bias[cg] : 0.0f;
  size_t row0 = (size_t)g*M + (size_t)wave*16 + mb;
  float v[8];
  #pragma unroll
  for (int r = 0; r < 8; ++r){
    v[r] = c[r] + bv;
    if (RELU) v[r] = fmaxf(v[r], 0.0f);
  }
  if (OUTMODE == 0){
    #pragma unroll
    for (int r = 0; r < 8; ++r) Obf[(row0 + r)*F + cg] = (bf16)v[r];
  } else if (OUTMODE == 1){
    unsigned int* Ot = Opk + (row0 >> 1)*F + cg;        // row0 even
    #pragma unroll
    for (int r = 0; r < 8; r += 2) Ot[(size_t)(r >> 1)*F] = pk2(v[r], v[r+1]);
  } else {
    #pragma unroll
    for (int r = 0; r < 8; ++r) Of[(row0 + r)*F + cg] = v[r];
  }
}

// double softmax over 25 clusters + ent_loss; writes s2t[c][p] (A-operand, row-major over c)
// and s2pk interleaved over p (B-operand for adj_pool)
__global__ void k_softmax2(const float* __restrict__ s, bf16* __restrict__ s2t,
                           bf16* __restrict__ s2pk, float* __restrict__ ent_acc){
  int i = blockIdx.x * blockDim.x + threadIdx.x;
  if (i >= NTOT) return;
  int g = i >> 6, p = i & 63;
  const float* r = s + (size_t)i*32;
  float v[25];
  float mx = -1e30f;
  #pragma unroll
  for (int c = 0; c < 25; ++c){ v[c] = r[c]; mx = fmaxf(mx, v[c]); }
  float se = 0.0f;
  #pragma unroll
  for (int c = 0; c < 25; ++c){ v[c] = __expf(v[c] - mx); se += v[c]; }
  float inv = 1.0f / se;
  #pragma unroll
  for (int c = 0; c < 25; ++c) v[c] *= inv;
  mx = -1e30f;
  #pragma unroll
  for (int c = 0; c < 25; ++c) mx = fmaxf(mx, v[c]);
  se = 0.0f;
  #pragma unroll
  for (int c = 0; c < 25; ++c){ v[c] = __expf(v[c] - mx); se += v[c]; }
  inv = 1.0f / se;
  bf16* T = s2t  + (size_t)g*2048;                       // [c*64 + p]
  bf16* Q = s2pk + (size_t)g*2048;                       // [(p>>1)*64 + 2c + (p&1)]
  float ent = 0.0f;
  #pragma unroll
  for (int c = 0; c < 25; ++c){
    float pr = v[c] * inv;
    ent += -pr * logf(pr + 1e-15f);
    bf16 hb = (bf16)pr;
    T[c*64 + p] = hb;
    Q[((p >> 1)*64) + 2*c + (p & 1)] = hb;
  }
  #pragma unroll
  for (int c = 25; c < 32; ++c){
    T[c*64 + p] = (bf16)0.0f;
    Q[((p >> 1)*64) + 2*c + (p & 1)] = (bf16)0.0f;
  }
  atomicAdd(ent_acc, ent * (1.0f / (4096.0f * 100.0f)));
}

// x_pool_g[32 x 128] = s2_g^T[32 x 64] @ h2_g[64 x 128]   (h2 packed)
__global__ void k_xpool(const bf16* __restrict__ s2t, const unsigned int* __restrict__ h2pk,
                        bf16* __restrict__ xp){
  int g = blockIdx.x, nt = blockIdx.y;                 // nt 0..7
  int lane = threadIdx.x & 31, wave = threadIdx.x >> 5;// wave 0..1 (M=32)
  const bf16* A = s2t + (size_t)g*2048 + (size_t)wave*16*64;
  const unsigned int* H = h2pk + (size_t)g*4096 + nt*16;
  v8f c = {};
  for (int k = 0; k < 64; k += 32){
    v16bf a = load_a(A + k, 64, lane);
    v16bf b = load_b_pk(H + (size_t)(k >> 1)*128, 128, lane);
    c = wmma_bf(a, b, c);
  }
  int n = lane & 15, mb = (lane >> 4) << 3;
  bf16* O = xp + (size_t)g*32*128 + (size_t)(wave*16)*128 + nt*16;
  #pragma unroll
  for (int r = 0; r < 8; ++r) O[(size_t)(mb + r)*128 + n] = (bf16)c[r];
}

// T1_g[32 x 64] = s2_g^T @ adjcnt_g   (adjcnt packed over K index)
__global__ void k_t1(const bf16* __restrict__ s2t, const unsigned int* __restrict__ adjpk,
                     bf16* __restrict__ t1){
  int g = blockIdx.x, nt = blockIdx.y;                 // nt 0..3
  int lane = threadIdx.x & 31, wave = threadIdx.x >> 5;
  const bf16* A = s2t + (size_t)g*2048 + (size_t)wave*16*64;
  const unsigned int* Bm = adjpk + (size_t)g*2048 + nt*16;
  v8f c = {};
  for (int k = 0; k < 64; k += 32){
    v16bf a = load_a(A + k, 64, lane);
    v16bf b = load_b_pk(Bm + (size_t)(k >> 1)*64, 64, lane);
    c = wmma_bf(a, b, c);
  }
  int n = lane & 15, mb = (lane >> 4) << 3;
  bf16* O = t1 + (size_t)g*32*64 + (size_t)(wave*16)*64 + nt*16;
  #pragma unroll
  for (int r = 0; r < 8; ++r) O[(size_t)(mb + r)*64 + n] = (bf16)c[r];
}

// adj_pool_g[32 x 32] = T1_g[32 x 64] @ s2_g[64 x 32]   (s2 packed; fp32 output)
__global__ void k_adjpool(const bf16* __restrict__ t1, const unsigned int* __restrict__ s2pk,
                          float* __restrict__ ap){
  int g = blockIdx.x, nt = blockIdx.y;                 // nt 0..1
  int lane = threadIdx.x & 31, wave = threadIdx.x >> 5;
  const bf16* A = t1 + (size_t)g*32*64 + (size_t)(wave*16)*64;
  const unsigned int* Bm = s2pk + (size_t)g*1024 + nt*16;
  v8f c = {};
  for (int k = 0; k < 64; k += 32){
    v16bf a = load_a(A + k, 64, lane);
    v16bf b = load_b_pk(Bm + (size_t)(k >> 1)*32, 32, lane);
    c = wmma_bf(a, b, c);
  }
  int n = lane & 15, mb = (lane >> 4) << 3;
  float* O = ap + (size_t)g*1024 + (size_t)(wave*16)*32 + nt*16;
  #pragma unroll
  for (int r = 0; r < 8; ++r) O[(size_t)(mb + r)*32 + n] = c[r];
}

// link_loss numerator: sum over 64x64 of (adj - s2 s2^T)^2 per graph
__global__ void k_link(const bf16* __restrict__ s2t, const bf16* __restrict__ adjcnt,
                       float* __restrict__ link_acc){
  int g = blockIdx.x, tid = threadIdx.x;
  __shared__ float S[64*25];
  __shared__ float bsum;
  for (int idx = tid; idx < 64*25; idx += 256){
    int c = idx >> 6, p = idx & 63;                     // S[p*25+c] layout below
    S[p*25 + c] = (float)s2t[(size_t)g*2048 + c*64 + p];
  }
  if (tid == 0) bsum = 0.0f;
  __syncthreads();
  const bf16* A = adjcnt + (size_t)g*4096;
  float local = 0.0f;
  for (int idx = tid; idx < 4096; idx += 256){
    int n = idx >> 6, m = idx & 63;
    float dot = 0.0f;
    #pragma unroll
    for (int c = 0; c < 25; ++c) dot += S[n*25 + c] * S[m*25 + c];
    float a = (float)A[((n >> 1)*128) + 2*m + (n & 1)]; // interleaved layout
    float d = a - dot;
    local += d*d;
  }
  atomicAdd(&bsum, local);
  __syncthreads();
  if (tid == 0) atomicAdd(link_acc, bsum);
}

// DenseGCNConv normalization: diag=1, d=max(rowsum,1)^-0.5, A_n = d_i A d_j (bf16, 32-padded)
__global__ void k_normadj(const float* __restrict__ ap, bf16* __restrict__ an){
  int g = blockIdx.x, i = threadIdx.x;   // block 32
  __shared__ float dsh[32];
  const float* P = ap + (size_t)g*1024;
  float d = 0.0f;
  if (i < 25){
    float rs = 0.0f;
    for (int j = 0; j < 25; ++j) rs += (j == i) ? 1.0f : P[i*32 + j];
    d = rsqrtf(fmaxf(rs, 1.0f));
  }
  dsh[i] = d;
  __syncthreads();
  bf16* O = an + (size_t)g*1024;
  for (int j = 0; j < 32; ++j){
    float a = (i < 25 && j < 25) ? ((j == i) ? 1.0f : P[i*32 + j]) : 0.0f;
    O[i*32 + j] = (bf16)(dsh[i]*dsh[j]*a);
  }
}

// out[g] = mean_{m<25} h4_g[m,:] @ Wl + bl
__global__ void k_final(const bf16* __restrict__ h4, const float* __restrict__ Wl,
                        const float* __restrict__ bl, float* __restrict__ out){
  int g = blockIdx.x, f = threadIdx.x;   // block 128
  __shared__ float mv[128];
  const bf16* H = h4 + (size_t)g*32*128;
  float s = 0.0f;
  for (int m = 0; m < 25; ++m) s += (float)H[(size_t)m*128 + f];
  mv[f] = s * (1.0f/25.0f);
  __syncthreads();
  if (f < 3){
    float acc = bl[f];
    for (int q = 0; q < 128; ++q) acc += mv[q] * Wl[q*3 + f];
    out[(size_t)g*3 + f] = acc;
  }
}

__global__ void k_fin(const float* __restrict__ scal, float* __restrict__ out){
  out[12288] = sqrtf(scal[0]) * (1.0f / (4096.0f * 100.0f * 100.0f));
  out[12289] = scal[1];
}

// ---- launch -----------------------------------------------------------------
extern "C" void kernel_launch(void* const* d_in, const int* in_sizes, int n_in,
                              void* d_out, int out_size, void* d_ws, size_t ws_size,
                              hipStream_t stream){
  const float* x   = (const float*)d_in[0];
  const int*   src = (const int*)d_in[1];
  const int*   dst = (const int*)d_in[2];
  const float* W1  = (const float*)d_in[4];
  const float* b1  = (const float*)d_in[5];
  const float* W2  = (const float*)d_in[6];
  const float* b2  = (const float*)d_in[7];
  const float* Wp  = (const float*)d_in[8];
  const float* bp  = (const float*)d_in[9];
  const float* W3  = (const float*)d_in[10];
  const float* b3  = (const float*)d_in[11];
  const float* W4  = (const float*)d_in[12];
  const float* b4  = (const float*)d_in[13];
  const float* Wl  = (const float*)d_in[14];
  const float* bl  = (const float*)d_in[15];
  float* out = (float*)d_out;

  size_t off = 0;
  auto carve = [&](size_t bytes) -> char* {
    char* p = (char*)d_ws + off;
    off += (bytes + 255) & ~(size_t)255;
    return p;
  };
  bf16*  adjcnt = (bf16*)carve((size_t)BGR*4096*2);            // 32 MB  counts, pair-interleaved
  bf16*  ahat   = (bf16*)carve((size_t)BGR*4096*2);            // 32 MB  normalized agg matrix
  unsigned int* buf1 = (unsigned int*)carve((size_t)NTOT*64*4);// 64 MB  X@W staging (packed pairs)
  bf16*  h1     = (bf16*)carve((size_t)NTOT*128*2);            // 64 MB  (later h3/h4)
  unsigned int* h2pk = (unsigned int*)carve((size_t)NTOT*64*4);// 64 MB  h2 (packed pairs)
  unsigned int* gppk = (unsigned int*)carve((size_t)NTOT*16*4);// 16 MB  (later A_n reuse)
  float* sbuf   = (float*)carve((size_t)NTOT*32*4);            // 32 MB  pre-softmax s
  bf16*  s2t    = (bf16*)carve((size_t)BGR*2048*2);            // 16 MB  s2^T  [g][c][p]
  bf16*  s2pk   = (bf16*)carve((size_t)BGR*2048*2);            // 16 MB  s2 packed pairs
  bf16*  xpool  = (bf16*)carve((size_t)BGR*32*128*2);          // 32 MB
  bf16*  t1     = (bf16*)carve((size_t)BGR*32*64*2);           // 16 MB
  float* apool  = (float*)carve((size_t)BGR*1024*4);           // 16 MB
  unsigned int* W1p = (unsigned int*)carve(64*128*4);
  unsigned int* W2p = (unsigned int*)carve(64*128*4);
  unsigned int* W3p = (unsigned int*)carve(64*128*4);
  unsigned int* W4p = (unsigned int*)carve(64*128*4);
  unsigned int* Wpp = (unsigned int*)carve(64*32*4);
  float* scal   = (float*)carve(256);                          // [0]=link, [1]=ent
  bf16*  an     = (bf16*)gppk;                                 // reuse (gp dead by then)
  bf16*  h3     = h1;                                          // reuse (h1 dead by then)
  bf16*  h4     = h1 + (size_t)BGR*32*128;

  k_zero<<<1, 32, 0, stream>>>(scal, 2);
  k_packW<<<32, 256, 0, stream>>>(W1, W1p, 128, 128, 128);
  k_packW<<<32, 256, 0, stream>>>(W2, W2p, 128, 128, 128);
  k_packW<<<32, 256, 0, stream>>>(W3, W3p, 128, 128, 128);
  k_packW<<<32, 256, 0, stream>>>(W4, W4p, 128, 128, 128);
  k_packW<<<8,  256, 0, stream>>>(Wp, Wpp, 128, 25, 32);
  k_adj<<<BGR, 256, 0, stream>>>(src, dst, adjcnt, ahat);

  // h1 = relu(Ahat @ (x @ W1) + b1)
  k_gemm_flat<float><<<dim3(4096, 8), 128, 0, stream>>>(x, W1p, buf1, 128, 128);
  k_graph_agg<true, 0><<<dim3(BGR, 8), 128, 0, stream>>>(ahat, buf1, b1, 128, h1, nullptr, nullptr, 64, 64, 128);
  // h2 = relu(Ahat @ (h1 @ W2) + b2)      (h2 stored packed for x_pool B-operand)
  k_gemm_flat<bf16><<<dim3(4096, 8), 128, 0, stream>>>(h1, W2p, buf1, 128, 128);
  k_graph_agg<true, 1><<<dim3(BGR, 8), 128, 0, stream>>>(ahat, buf1, b2, 128, nullptr, h2pk, nullptr, 64, 64, 128);
  // s = Ahat @ (h1 @ Wp) + bp   (fp32, 32-col padded)
  k_gemm_flat<bf16><<<dim3(4096, 2), 128, 0, stream>>>(h1, Wpp, gppk, 128, 32);
  k_graph_agg<false, 2><<<dim3(BGR, 2), 128, 0, stream>>>(ahat, gppk, bp, 25, nullptr, nullptr, sbuf, 64, 64, 32);

  k_softmax2<<<NTOT/256, 256, 0, stream>>>(sbuf, s2t, s2pk, scal + 1);

  k_xpool<<<dim3(BGR, 8), 64, 0, stream>>>(s2t, h2pk, xpool);
  k_t1<<<dim3(BGR, 4), 64, 0, stream>>>(s2t, (const unsigned int*)adjcnt, t1);
  k_adjpool<<<dim3(BGR, 2), 64, 0, stream>>>(t1, (const unsigned int*)s2pk, apool);
  k_link<<<BGR, 256, 0, stream>>>(s2t, adjcnt, scal);
  k_normadj<<<BGR, 32, 0, stream>>>(apool, an);

  // h3 = relu(A_n @ (x_pool @ W3) + b3)
  k_gemm_flat<bf16><<<dim3(2048, 8), 128, 0, stream>>>(xpool, W3p, buf1, 128, 128);
  k_graph_agg<true, 0><<<dim3(BGR, 8), 64, 0, stream>>>(an, buf1, b3, 128, h3, nullptr, nullptr, 32, 32, 128);
  // h4 = relu(A_n @ (h3 @ W4) + b4)
  k_gemm_flat<bf16><<<dim3(2048, 8), 128, 0, stream>>>(h3, W4p, buf1, 128, 128);
  k_graph_agg<true, 0><<<dim3(BGR, 8), 64, 0, stream>>>(an, buf1, b4, 128, h4, nullptr, nullptr, 32, 32, 128);

  k_final<<<BGR, 128, 0, stream>>>(h4, Wl, bl, out);
  k_fin<<<1, 1, 0, stream>>>(scal, out);

  (void)in_sizes; (void)n_in; (void)out_size; (void)ws_size;
}
